// DecoderRNN_81707457839118
// MI455X (gfx1250) — matrix operand order repaired and checked by reading, not compiled, not run on hardware
//
#include <hip/hip_runtime.h>
#include <hip/hip_bf16.h>
#include <stdint.h>

// ---------------------------------------------------------------------------
// DecoderRNN on MI455X (gfx1250): bf16 WMMA GEMMs with LDS staging via
// GLOBAL_LOAD_ASYNC_TO_LDS_B128, triple-buffered, one barrier per 64-wide
// K chunk; fragment ds_loads batched ahead of the WMMA burst.
// Weights converted once to bf16 and resident in the 192MB L2.
// ---------------------------------------------------------------------------
static constexpr int kB    = 512;    // batch
static constexpr int kH    = 2048;   // hidden
static constexpr int kOut  = 165;    // J*D
static constexpr int kNcls = 40;
static constexpr int kLen  = 240;
static constexpr int kXPad = 352;    // 2*kOut=330 padded to mult of 32
static constexpr int kKw   = 2432;   // padded recurrent K (= 38 * 64)
static constexpr int kG4   = 4 * kH; // 8192 gate width
static constexpr int kKIn  = 192;    // kOut padded (init GEMMs, = 3 * 64)
static constexpr int kNfc2 = 256;    // fc2 rows padded to block tile

static constexpr int kKC    = 64;    // K chunk per LDS stage
static constexpr int kPitch = 144;   // LDS row pitch bytes (128B data + 16B pad)

typedef __attribute__((ext_vector_type(16))) __bf16 v16bf;
typedef __attribute__((ext_vector_type(8)))  float  v8f;

union Frag16 { uint4 q[2]; v16bf v; };

__device__ __forceinline__ unsigned short f2bf(float x) {
    union { float f; uint32_t u; } c; c.f = x;
    uint32_t u = c.u;
    u += 0x7FFFu + ((u >> 16) & 1u);          // round-to-nearest-even
    return (unsigned short)(u >> 16);
}

// 16B global -> LDS async copy (ASYNCcnt). ldsOff: byte offset into dynamic LDS.
__device__ __forceinline__ void async_cp16(unsigned ldsOff, const void* g) {
    asm volatile("global_load_async_to_lds_b128 %0, %1, off"
                 :: "v"(ldsOff), "v"(g) : "memory");
}

// ---------------------------------------------------------------------------
// bf16 WMMA GEMM:  C[M,N] = A[M,K] @ W[N,K]^T (+bias) (opt ReLU)
// Block tile (32*MW) x 256, 8 waves (2 x 4), wave tile (16*MW) x 64.
// K chunked by 64, staged in LDS (A: 32*MW rows, B: 256 rows, 144B pitch),
// triple buffered; async copies run 2 chunks ahead; 1 barrier per chunk.
// Requires: M % (32*MW) == 0, K % 64 == 0, K >= 128, W zero-padded to
// grid.x*256 rows, lda/ldw multiples of 8.
// ---------------------------------------------------------------------------
template <int MW>
struct FragSet {
    Frag16 a[MW];
    Frag16 b[4];
};

template <int MW>
__device__ __forceinline__ void mma_frags(v8f (&acc)[MW][4], const FragSet<MW>& f)
{
#pragma unroll
    for (int i = 0; i < MW; ++i)
#pragma unroll
        for (int t = 0; t < 4; ++t)
            acc[i][t] = __builtin_amdgcn_wmma_f32_16x16x32_bf16(
                false, f.a[i].v, false, f.b[t].v, (short)0, acc[i][t], false, false);
}

// Issue the (8 + MW) async copy instructions (per wave) for one 64-wide chunk.
// B tile: 256 rows x 128B  (8 passes x 256 threads x 16B)
// A tile: 32*MW rows x 128B (MW passes)
template <int MW>
__device__ __forceinline__ void copy_chunk(const unsigned short* Ablk, int lda,
                                           const unsigned short* Wblk, int ldw,
                                           int k0, unsigned stBase)
{
    const unsigned aBytes = (unsigned)(32 * MW) * kPitch;
    const int tid = threadIdx.x;
#pragma unroll
    for (int p = 0; p < 8; ++p) {
        int sidx = tid + p * 256;            // 0..2047
        int row = sidx >> 3, seg = sidx & 7;
        async_cp16(stBase + aBytes + (unsigned)row * kPitch + seg * 16,
                   Wblk + (size_t)row * ldw + k0 + seg * 8);
    }
#pragma unroll
    for (int p = 0; p < MW; ++p) {
        int sidx = tid + p * 256;            // 0..256*MW-1
        int row = sidx >> 3, seg = sidx & 7;
        async_cp16(stBase + (unsigned)row * kPitch + seg * 16,
                   Ablk + (size_t)row * lda + k0 + seg * 8);
    }
}

// ds_load one FragSet for 32-wide sub-step ks (0 or 1) of the staged chunk.
template <int MW>
__device__ __forceinline__ void load_lds_frags(FragSet<MW>& f, const char* smem,
                                               unsigned stBase, int ks,
                                               int wave, int half, int l15)
{
    const unsigned aBase = stBase;
    const unsigned bBase = stBase + (unsigned)(32 * MW) * kPitch;
#pragma unroll
    for (int i = 0; i < MW; ++i) {
        // A frag row (waveM*16*MW + i*16 + l15); elems 0-7 @ K=half*8, 8-15 @ +16
        unsigned ao = aBase + (unsigned)((wave & 1) * (16 * MW) + i * 16 + l15) * kPitch
                    + ks * 64 + half * 16;
        f.a[i].q[0] = *(const uint4*)(smem + ao);
        f.a[i].q[1] = *(const uint4*)(smem + ao + 32);
    }
#pragma unroll
    for (int t = 0; t < 4; ++t) {
        // B frag: weight row ((wave>>1)*64 + t*16 + l15); elems 0-7 @ K=half*8, 8-15 @ +16
        unsigned bo = bBase + (unsigned)((wave >> 1) * 64 + t * 16 + l15) * kPitch
                    + ks * 64 + half * 16;
        f.b[t].q[0] = *(const uint4*)(smem + bo);
        f.b[t].q[1] = *(const uint4*)(smem + bo + 32);
    }
}

// Batch ALL ds_loads of the chunk, then run the 8*MW WMMAs: in-order DS
// returns give progressive partial s_wait_dscnt before each WMMA.
template <int MW>
__device__ __forceinline__ void compute_chunk(v8f (&acc)[MW][4], const char* smem,
                                              unsigned stBase, int wave, int half, int l15)
{
    FragSet<MW> f0, f1;
    load_lds_frags<MW>(f0, smem, stBase, 0, wave, half, l15);
    load_lds_frags<MW>(f1, smem, stBase, 1, wave, half, l15);
    mma_frags<MW>(acc, f0);
    mma_frags<MW>(acc, f1);
}

template <int MW>
__global__ __launch_bounds__(256) void gemm_bf16_wmma(
    const unsigned short* __restrict__ A, int lda,
    const unsigned short* __restrict__ W, int ldw,
    const float* __restrict__ bias,
    float* __restrict__ outF, int ldcF,
    unsigned short* __restrict__ outB, int ldcB,
    int K, int Nreal, int relu)
{
    extern __shared__ char smem[];
    const unsigned stage = (unsigned)(32 * MW + 256) * kPitch;

    const int lane = threadIdx.x & 31;
    const int wave = threadIdx.x >> 5;
    const int half = lane >> 4;
    const int l15  = lane & 15;
    const int m0 = blockIdx.y * (32 * MW) + (wave & 1) * (16 * MW);
    const int n0 = blockIdx.x * 256 + (wave >> 1) * 64;

    const unsigned short* Ablk = A + (size_t)(blockIdx.y * (32 * MW)) * lda;
    const unsigned short* Wblk = W + (size_t)(blockIdx.x * 256) * ldw;

    v8f acc[MW][4];
#pragma unroll
    for (int i = 0; i < MW; ++i)
#pragma unroll
        for (int t = 0; t < 4; ++t)
            acc[i][t] = (v8f){};

    const int nch = K / kKC;                 // >= 2
    copy_chunk<MW>(Ablk, lda, Wblk, ldw, 0, 0);
    copy_chunk<MW>(Ablk, lda, Wblk, ldw, kKC, stage);

    unsigned stC = 0;                        // compute stage byte offset
    unsigned stI = 2 * stage;                // issue stage byte offset
    for (int c = 0; c < nch; ++c) {
        // Own chunk-c copies complete (chunk c+1's 8+MW may stay in flight).
        if (c + 1 < nch) {
            if constexpr (MW == 1) asm volatile("s_wait_asynccnt 9"  ::: "memory");
            else                   asm volatile("s_wait_asynccnt 10" ::: "memory");
        } else {
            asm volatile("s_wait_asynccnt 0" ::: "memory");
        }
        // Barrier: (a) all waves' chunk-c copies landed; (b) all waves finished
        // reading stage stI (= compute stage of chunk c-1) -> safe to overwrite.
        __syncthreads();
        compute_chunk<MW>(acc, smem, stC, wave, half, l15);
        if (c + 2 < nch)
            copy_chunk<MW>(Ablk, lda, Wblk, ldw, (c + 2) * kKC, stI);
        stC += stage; if (stC == 3 * stage) stC = 0;
        stI += stage; if (stI == 3 * stage) stI = 0;
    }

#pragma unroll
    for (int t = 0; t < 4; ++t) {
        int n = n0 + t * 16 + l15;
        if (n >= Nreal) continue;
        float bs = bias ? bias[n] : 0.0f;
#pragma unroll
        for (int i = 0; i < MW; ++i) {
#pragma unroll
            for (int v = 0; v < 8; ++v) {
                float val = acc[i][t][v] + bs;
                if (relu) val = fmaxf(val, 0.0f);
                int m = m0 + i * 16 + half * 8 + v;
                if (outF) outF[(size_t)m * ldcF + n] = val;
                if (outB) outB[(size_t)m * ldcB + n] = f2bf(val);
            }
        }
    }
}

template <int MW>
static inline unsigned lds_bytes() { return (unsigned)(32 * MW + 256) * kPitch * 3; }

// --------------------------- elementwise LSTM cell -------------------------
__global__ void lstm_cell(const float* __restrict__ gates,
                          float* __restrict__ c,
                          unsigned short* __restrict__ xh)
{
    int idx = blockIdx.x * blockDim.x + threadIdx.x;
    if (idx >= kB * kH) return;
    int b = idx / kH, u = idx - b * kH;
    const float* g = gates + (size_t)b * kG4;
    float gi = g[u], gf = g[kH + u], gg = g[2 * kH + u], go = g[3 * kH + u];
    float si = 1.0f / (1.0f + __expf(-gi));
    float sf = 1.0f / (1.0f + __expf(-gf));
    float so = 1.0f / (1.0f + __expf(-go));
    float cc = sf * c[idx] + si * tanhf(gg);
    c[idx] = cc;
    float h = so * tanhf(cc);
    xh[(size_t)b * kKw + kXPad + u] = f2bf(h);   // h feeds next GEMMs as bf16
}

// ------------------- fp32 -> padded raw-bf16 weight convert ----------------
__global__ void cvt_pad(unsigned short* __restrict__ dst, int dld, int doff,
                        const float* __restrict__ src, int sld,
                        int Nspan, int Nreal, int Kspan, int Kreal)
{
    int idx = blockIdx.x * blockDim.x + threadIdx.x;
    if (idx >= Nspan * Kspan) return;
    int n = idx / Kspan, k = idx - n * Kspan;
    float v = (n < Nreal && k < Kreal) ? src[(size_t)n * sld + k] : 0.0f;
    dst[(size_t)n * dld + doff + k] = f2bf(v);
}

__global__ void add_bias2(float* __restrict__ dst, const float* __restrict__ a,
                          const float* __restrict__ b, int n)
{
    int i = blockIdx.x * blockDim.x + threadIdx.x;
    if (i < n) dst[i] = a[i] + b[i];
}

// xh[:, 0:165]=frame0 ; [165:330]=enc ; [330:352]=0 ; tail [2400:2432]=0
__global__ void init_x(unsigned short* __restrict__ xh,
                       const float* __restrict__ inputs,
                       const int* __restrict__ labels,
                       const float* __restrict__ W_enc,
                       const float* __restrict__ b_enc)
{
    int idx = blockIdx.x * blockDim.x + threadIdx.x;
    if (idx >= kB * (kXPad + 32)) return;
    int b = idx / (kXPad + 32), col = idx - b * (kXPad + 32);
    float v = 0.0f;
    int dstcol = col;
    if (col < kOut) {
        v = inputs[(size_t)b * kOut + col];
    } else if (col < 2 * kOut) {
        int j = col - kOut;
        v = W_enc[(size_t)j * kNcls + labels[b]] + b_enc[j];
    } else if (col >= kXPad) {
        dstcol = kKw - 32 + (col - kXPad);     // tail zeros
    }
    xh[(size_t)b * kKw + dstcol] = f2bf(v);
}

// frame0 as bf16, K padded 165 -> 192 (A-matrix of the h0/c0 init GEMMs)
__global__ void init_f0(unsigned short* __restrict__ f0,
                        const float* __restrict__ inputs)
{
    int idx = blockIdx.x * blockDim.x + threadIdx.x;
    if (idx >= kB * kKIn) return;
    int b = idx / kKIn, k = idx - b * kKIn;
    f0[idx] = f2bf(k < kOut ? inputs[(size_t)b * kOut + k] : 0.0f);
}

// ---------------------------------------------------------------------------
extern "C" void kernel_launch(void* const* d_in, const int* in_sizes, int n_in,
                              void* d_out, int out_size, void* d_ws, size_t ws_size,
                              hipStream_t stream)
{
    const float* inputs = (const float*)d_in[0];
    const int*   labels = (const int*)  d_in[1];
    // d_in[2] = length (fixed 240 in reference)
    const float* W_enc = (const float*)d_in[3];
    const float* b_enc = (const float*)d_in[4];
    const float* W_ih  = (const float*)d_in[5];
    const float* b_ih  = (const float*)d_in[6];
    const float* W_hh  = (const float*)d_in[7];
    const float* b_hh  = (const float*)d_in[8];
    const float* W_fc1 = (const float*)d_in[9];
    const float* b_fc1 = (const float*)d_in[10];
    const float* W_fc2 = (const float*)d_in[11];
    const float* b_fc2 = (const float*)d_in[12];
    const float* W_inh = (const float*)d_in[13];
    const float* b_inh = (const float*)d_in[14];
    const float* W_inc = (const float*)d_in[15];
    const float* b_inc = (const float*)d_in[16];
    float* out = (float*)d_out;          // [B, LEN, OUT] f32

    // ---- workspace carve-out (~77 MB) ----
    char* ws = (char*)d_ws;
    size_t off = 0;
    auto carve = [&](size_t bytes) -> void* {
        void* p = ws + off;
        off += (bytes + 255) & ~(size_t)255;
        return p;
    };
    unsigned short* Wcat  = (unsigned short*)carve((size_t)kG4  * kKw * 2);   // [8192,2432]
    unsigned short* Wfc1b = (unsigned short*)carve((size_t)kH   * kH  * 2);   // [2048,2048]
    unsigned short* Wfc2b = (unsigned short*)carve((size_t)kNfc2* kH  * 2);   // [256, 2048]
    unsigned short* Winhb = (unsigned short*)carve((size_t)kH   * kKIn * 2);  // [2048, 192]
    unsigned short* Wincb = (unsigned short*)carve((size_t)kH   * kKIn * 2);
    float*          biasg = (float*)         carve((size_t)kG4 * 4);          // b_ih+b_hh
    unsigned short* xh    = (unsigned short*)carve((size_t)kB * kKw * 2);     // [B,2432]
    float*          gates = (float*)         carve((size_t)kB * kG4 * 4);     // [B,8192]
    float*          cbuf  = (float*)         carve((size_t)kB * kH * 4);      // [B,2048]
    unsigned short* rbuf  = (unsigned short*)carve((size_t)kB * kH * 2);      // [B,2048]
    unsigned short* f0    = (unsigned short*)carve((size_t)kB * kKIn * 2);    // [B,192]
    if (off > ws_size) return;  // scratch too small: bail deterministically

    auto g1 = [](long n) { return dim3((unsigned)((n + 255) / 256)); };

    // ---- one-time (per call) weight conversion to L2-resident bf16 ----
    cvt_pad<<<g1((long)kG4 * kXPad), 256, 0, stream>>>(
        Wcat, kKw, 0,     W_ih, 2 * kOut, kG4, kG4, kXPad, 2 * kOut);
    cvt_pad<<<g1((long)kG4 * (kKw - kXPad)), 256, 0, stream>>>(
        Wcat, kKw, kXPad, W_hh, kH,       kG4, kG4, kKw - kXPad, kH);
    cvt_pad<<<g1((long)kH * kH), 256, 0, stream>>>(
        Wfc1b, kH, 0,     W_fc1, kH,      kH,  kH,  kH,   kH);
    cvt_pad<<<g1((long)kNfc2 * kH), 256, 0, stream>>>(
        Wfc2b, kH, 0,     W_fc2, kH,      kNfc2, kOut, kH, kH);
    cvt_pad<<<g1((long)kH * kKIn), 256, 0, stream>>>(
        Winhb, kKIn, 0,   W_inh, kOut,    kH,  kH,  kKIn, kOut);
    cvt_pad<<<g1((long)kH * kKIn), 256, 0, stream>>>(
        Wincb, kKIn, 0,   W_inc, kOut,    kH,  kH,  kKIn, kOut);
    add_bias2<<<g1(kG4), 256, 0, stream>>>(biasg, b_ih, b_hh, kG4);

    // ---- initial state: xh = [frame0 | enc | 0 | h0 | 0], c = c0 ----
    init_x<<<g1((long)kB * (kXPad + 32)), 256, 0, stream>>>(xh, inputs, labels, W_enc, b_enc);
    init_f0<<<g1((long)kB * kKIn), 256, 0, stream>>>(f0, inputs);
    // h0 = frame0 @ W_inh.T + b_inh  -> bf16 into xh[:, 352:]
    gemm_bf16_wmma<1><<<dim3(kH / 256, kB / 32), 256, lds_bytes<1>(), stream>>>(
        f0, kKIn, Winhb, kKIn, b_inh,
        (float*)nullptr, 0, xh + kXPad, kKw, kKIn, kH, 0);
    // c0 = frame0 @ W_inc.T + b_inc  -> f32 into cbuf
    gemm_bf16_wmma<1><<<dim3(kH / 256, kB / 32), 256, lds_bytes<1>(), stream>>>(
        f0, kKIn, Wincb, kKIn, b_inc,
        cbuf, kH, (unsigned short*)nullptr, 0, kKIn, kH, 0);

    // ---- sequential recurrence: 4 kernels per timestep ----
    for (int t = 0; t < kLen; ++t) {
        // gates = [x|h] @ Wcat.T + (b_ih + b_hh)     [512 x 8192], K=2432
        gemm_bf16_wmma<2><<<dim3(kG4 / 256, kB / 64), 256, lds_bytes<2>(), stream>>>(
            xh, kKw, Wcat, kKw, biasg,
            gates, kG4, (unsigned short*)nullptr, 0, kKw, kG4, 0);
        // LSTM cell: updates cbuf, writes h (bf16) into xh[:, 352:]
        lstm_cell<<<g1((long)kB * kH), 256, 0, stream>>>(gates, cbuf, xh);
        // r = relu(h @ W_fc1.T + b_fc1) -> bf16       [512 x 2048]
        gemm_bf16_wmma<1><<<dim3(kH / 256, kB / 32), 256, lds_bytes<1>(), stream>>>(
            xh + kXPad, kKw, Wfc1b, kH, b_fc1,
            (float*)nullptr, 0, rbuf, kH, kH, kH, 1);
        // out_t = r @ W_fc2.T + b_fc2 -> f32 d_out[:,t,:] and bf16 xh[:,0:165]
        gemm_bf16_wmma<1><<<dim3(kNfc2 / 256, kB / 32), 256, lds_bytes<1>(), stream>>>(
            rbuf, kH, Wfc2b, kH, b_fc2,
            out + (size_t)t * kOut, kLen * kOut, xh, kKw, kH, kOut, 0);
    }
}
